// AgentLayer_29137058136160
// MI455X (gfx1250) — matrix operand order repaired and checked by reading, not compiled, not run on hardware
//
#include <hip/hip_runtime.h>
#include <hip/hip_bf16.h>

#define BB 512
#define TT 256
#define DD 256
#define HH 128
#define SS 64
#define AAC 10
#define UU 64
#define TD (TT*DD)

typedef __attribute__((ext_vector_type(16))) __bf16 v16bf;
typedef __attribute__((ext_vector_type(8)))  __bf16 v8bf;
typedef __attribute__((ext_vector_type(8)))  float  v8f;

// bf16 workspace layout (element offsets, all 16B-aligned)
#define OFF_WIH   0          // [384][256]
#define OFF_WHH   98304      // [384][128]
#define OFF_A1W1  147456     // [64][192]
#define OFF_A2W1  159744     // [64][320]
#define OFF_A1W2  180224     // [16][64] padded (rows 10..15 zero)
#define OFF_A2W2  181248     // [16][64] padded
#define OFF_INITW 182272     // [128][64]
#define OFF_FUSW  190464     // [128][192]
#define WS_TOTAL  215040

__global__ void prep_weights(const float* __restrict__ wih, const float* __restrict__ whh,
                             const float* __restrict__ a1w1, const float* __restrict__ a2w1,
                             const float* __restrict__ a1w2, const float* __restrict__ a2w2,
                             const float* __restrict__ initw, const float* __restrict__ fusw,
                             __bf16* __restrict__ ws) {
  int i = blockIdx.x * blockDim.x + threadIdx.x;
  if (i >= WS_TOTAL) return;
  float v;
  if (i < OFF_WHH)        v = wih[i];
  else if (i < OFF_A1W1)  v = whh[i - OFF_WHH];
  else if (i < OFF_A2W1)  v = a1w1[i - OFF_A1W1];
  else if (i < OFF_A1W2)  v = a2w1[i - OFF_A2W1];
  else if (i < OFF_A2W2)  { int d = i - OFF_A1W2; int r = d >> 6, c = d & 63; v = (r < AAC) ? a1w2[r*UU + c] : 0.f; }
  else if (i < OFF_INITW) { int d = i - OFF_A2W2; int r = d >> 6, c = d & 63; v = (r < AAC) ? a2w2[r*UU + c] : 0.f; }
  else if (i < OFF_FUSW)  v = initw[i - OFF_INITW];
  else                    v = fusw[i - OFF_FUSW];
  ws[i] = (__bf16)v;
}

__device__ __forceinline__ v8f vzero8() {
  v8f z;
#pragma unroll
  for (int i = 0; i < 8; ++i) z[i] = 0.f;
  return z;
}

__device__ __forceinline__ v16bf combine16(v8bf lo, v8bf hb) {
  v16bf r;
#pragma unroll
  for (int i = 0; i < 8; ++i) { r[i] = lo[i]; r[i + 8] = hb[i]; }
  return r;
}

// A fragment (16x32 bf16, row-major source with leading dim ld): per ISA table,
// lane (m = l&15, hi = l>>4) holds K = k0 + hi*8 + e (e<8) and K = k0+16+hi*8+(e-8).
__device__ __forceinline__ v16bf afrag_lds(const __bf16* buf, int ld, int m, int k0, int hi) {
  const __bf16* p = buf + m * ld + k0;
  v8bf lo = *(const v8bf*)(p + hi * 8);
  v8bf hb = *(const v8bf*)(p + 16 + hi * 8);
  return combine16(lo, hb);
}

// Same A layout, sourced from row-major fp32 global memory (converted to bf16 in regs).
__device__ __forceinline__ v16bf afrag_g32(const float* base, size_t rowStride, int m, int k0, int hi) {
  const float* rp = base + (size_t)m * rowStride + k0;
  const float4* p1 = (const float4*)(rp + hi * 8);
  const float4* p2 = (const float4*)(rp + 16 + hi * 8);
  float4 q0 = p1[0], q1 = p1[1], q2 = p2[0], q3 = p2[1];
  v16bf r;
  r[0]=(__bf16)q0.x;  r[1]=(__bf16)q0.y;  r[2]=(__bf16)q0.z;  r[3]=(__bf16)q0.w;
  r[4]=(__bf16)q1.x;  r[5]=(__bf16)q1.y;  r[6]=(__bf16)q1.z;  r[7]=(__bf16)q1.w;
  r[8]=(__bf16)q2.x;  r[9]=(__bf16)q2.y;  r[10]=(__bf16)q2.z; r[11]=(__bf16)q2.w;
  r[12]=(__bf16)q3.x; r[13]=(__bf16)q3.y; r[14]=(__bf16)q3.z; r[15]=(__bf16)q3.w;
  return r;
}

// B fragment (32x16 bf16): lane (n = l&15, hi = l>>4) holds K = k0 + hi*8 + e (e<8)
// and K = k0+16+hi*8+(e-8), read from row-major weight W[N][K].
__device__ __forceinline__ v16bf bfrag(const __bf16* W, int ldk, int n0, int k0, int n, int hi) {
  const __bf16* p = W + (size_t)(n0 + n) * ldk + k0;
  v8bf lo = *(const v8bf*)(p + hi * 8);
  v8bf hb = *(const v8bf*)(p + 16 + hi * 8);
  return combine16(lo, hb);
}

__device__ __forceinline__ v8f wmma_bf(v16bf a, v16bf b, v8f c) {
  return __builtin_amdgcn_wmma_f32_16x16x32_bf16(false, a, false, b, (short)0, c, false, false);
}

__device__ __forceinline__ float sigmoidf_(float x) { return 1.0f / (1.0f + __expf(-x)); }

__global__ void __launch_bounds__(256, 1)
agent_scan(const float* __restrict__ x, const float* __restrict__ stat,
           const float* __restrict__ a1_b1, const float* __restrict__ a2_b1,
           const float* __restrict__ a1_b2, const float* __restrict__ a2_b2,
           const float* __restrict__ gru_bih, const float* __restrict__ gru_bhh,
           const float* __restrict__ init_b, const float* __restrict__ fus_b,
           const __bf16* __restrict__ ws, float* __restrict__ out) {
  const int tid  = threadIdx.x;
  const int wave = tid >> 5;
  const int lane = tid & 31;
  const int lm   = lane & 15;
  const int hi   = lane >> 4;
  const int b0   = blockIdx.x * 16;

  const __bf16* w_wih  = ws + OFF_WIH;
  const __bf16* w_whh  = ws + OFF_WHH;
  const __bf16* w_a1w1 = ws + OFF_A1W1;
  const __bf16* w_a2w1 = ws + OFF_A2W1;
  const __bf16* w_a1w2 = ws + OFF_A1W2;
  const __bf16* w_a2w2 = ws + OFF_A2W2;
  const __bf16* w_init = ws + OFF_INITW;
  const __bf16* w_fus  = ws + OFF_FUSW;

  __shared__ __align__(16) __bf16 ring[AAC][16][HH];   // 40 KB obs history
  __shared__ __align__(16) __bf16 h_bf[16][HH];        // h_{t-1}
  __shared__ __align__(16) __bf16 mean_bf[16][HH];     // obs mean
  __shared__ __align__(16) __bf16 wgt_bf[16][HH];      // weighted_h
  __shared__ __align__(16) __bf16 z1a[16][UU];
  __shared__ __align__(16) __bf16 z1b[16][UU];
  __shared__ float lg1[16][16];
  __shared__ float lg2[16][16];
  __shared__ int   idx_s[32];

  // zero the history ring (matches obs0 = zeros)
  for (int e = tid; e < AAC * 16 * HH; e += 256) ((__bf16*)ring)[e] = (__bf16)0.f;

  // per-lane loop-invariant biases (wave w owns gate columns j = 16w + lm)
  const int jc = 16 * wave + lm;
  const float c_bihr = gru_bih[jc],        c_bhhr = gru_bhh[jc];
  const float c_bihz = gru_bih[HH + jc],   c_bhhz = gru_bhh[HH + jc];
  const float c_bihn = gru_bih[2*HH + jc], c_bhhn = gru_bhh[2*HH + jc];
  const float c_initb = init_b[jc];
  const float c_fusb  = fus_b[jc];
  const float c_a1b1 = (wave < 4) ? a1_b1[16 * wave + lm] : 0.f;
  const float c_a2b1 = (wave >= 4) ? a2_b1[16 * (wave - 4) + lm] : 0.f;
  const float c_a1b2 = (wave == 0 && lm < AAC) ? a1_b2[lm] : 0.f;
  const float c_a2b2 = (wave == 1 && lm < AAC) ? a2_b2[lm] : 0.f;

  const float* statb = stat + (size_t)b0 * SS;
  __syncthreads();

  int wptr = 0;
#pragma unroll 1
  for (int t = 0; t < TT; ++t) {
    const float* xb = x + (size_t)b0 * TD + (size_t)t * DD;

    if (t == 0) {
      // weighted_h := cur_h = static @ init_w^T + init_b
      v8f acc = vzero8();
#pragma unroll
      for (int k0 = 0; k0 < SS; k0 += 32) {
        v16bf a = afrag_g32(statb, SS, lm, k0, hi);
        acc = wmma_bf(a, bfrag(w_init, SS, 16 * wave, k0, lm, hi), acc);
      }
#pragma unroll
      for (int v = 0; v < 8; ++v)
        wgt_bf[v + 8 * hi][jc] = (__bf16)(acc[v] + c_initb);
      __syncthreads();
    } else {
      // obs mean (ring already holds obs after shift: newest appended last step)
      for (int e = tid; e < 16 * HH; e += 256) {
        int m = e >> 7, j = e & 127;
        float s = 0.f;
#pragma unroll
        for (int q = 0; q < AAC; ++q) s += (float)ring[q][m][j];
        mean_bf[m][j] = (__bf16)(s * 0.1f);
      }
      __syncthreads();

      // action MLP layer 1 (waves 0-3: obs1 net; waves 4-7: obs2 net)
      {
        v8f acc = vzero8();
        if (wave < 4) {
#pragma unroll
          for (int k0 = 0; k0 < HH; k0 += 32) {
            v16bf a = afrag_lds(&mean_bf[0][0], HH, lm, k0, hi);
            acc = wmma_bf(a, bfrag(w_a1w1, HH + SS, 16 * wave, k0, lm, hi), acc);
          }
#pragma unroll
          for (int k0 = HH; k0 < HH + SS; k0 += 32) {
            v16bf a = afrag_g32(statb, SS, lm, k0 - HH, hi);
            acc = wmma_bf(a, bfrag(w_a1w1, HH + SS, 16 * wave, k0, lm, hi), acc);
          }
#pragma unroll
          for (int v = 0; v < 8; ++v)
            z1a[v + 8 * hi][16 * wave + lm] = (__bf16)tanhf(acc[v] + c_a1b1);
        } else {
          const int w2 = wave - 4;
#pragma unroll
          for (int k0 = 0; k0 < DD; k0 += 32) {
            v16bf a = afrag_g32(xb, TD, lm, k0, hi);
            acc = wmma_bf(a, bfrag(w_a2w1, DD + SS, 16 * w2, k0, lm, hi), acc);
          }
#pragma unroll
          for (int k0 = DD; k0 < DD + SS; k0 += 32) {
            v16bf a = afrag_g32(statb, SS, lm, k0 - DD, hi);
            acc = wmma_bf(a, bfrag(w_a2w1, DD + SS, 16 * w2, k0, lm, hi), acc);
          }
#pragma unroll
          for (int v = 0; v < 8; ++v)
            z1b[v + 8 * hi][16 * w2 + lm] = (__bf16)tanhf(acc[v] + c_a2b1);
        }
      }
      __syncthreads();

      // layer 2 -> logits
      if (wave == 0) {
        v8f acc = vzero8();
#pragma unroll
        for (int k0 = 0; k0 < UU; k0 += 32)
          acc = wmma_bf(afrag_lds(&z1a[0][0], UU, lm, k0, hi),
                        bfrag(w_a1w2, UU, 0, k0, lm, hi), acc);
#pragma unroll
        for (int v = 0; v < 8; ++v) lg1[v + 8 * hi][lm] = acc[v] + c_a1b2;
      } else if (wave == 1) {
        v8f acc = vzero8();
#pragma unroll
        for (int k0 = 0; k0 < UU; k0 += 32)
          acc = wmma_bf(afrag_lds(&z1b[0][0], UU, lm, k0, hi),
                        bfrag(w_a2w2, UU, 0, k0, lm, hi), acc);
#pragma unroll
        for (int v = 0; v < 8; ++v) lg2[v + 8 * hi][lm] = acc[v] + c_a2b2;
      }
      __syncthreads();

      // argmax over A=10 (first-max like jnp.argmax)
      if (tid < 16) {
        int best = 0; float bv = lg1[tid][0];
#pragma unroll
        for (int c = 1; c < AAC; ++c) { float v = lg1[tid][c]; if (v > bv) { bv = v; best = c; } }
        idx_s[tid] = best;
      } else if (tid < 32) {
        int r = tid - 16, best = 0; float bv = lg2[r][0];
#pragma unroll
        for (int c = 1; c < AAC; ++c) { float v = lg2[r][c]; if (v > bv) { bv = v; best = c; } }
        idx_s[tid] = best;
      }
      __syncthreads();

      // gather + weighted_h = 0.25*(h1+h2) + 0.5*cur_h   (logical idx -> physical ring slot)
      for (int e = tid; e < 16 * HH; e += 256) {
        int m = e >> 7, j = e & 127;
        int p1 = wptr + idx_s[m];      if (p1 >= AAC) p1 -= AAC;
        int p2 = wptr + idx_s[16 + m]; if (p2 >= AAC) p2 -= AAC;
        float wv = 0.25f * ((float)ring[p1][m][j] + (float)ring[p2][m][j])
                 + 0.5f * (float)h_bf[m][j];
        wgt_bf[m][j] = (__bf16)wv;
      }
      __syncthreads();
    }

    // ---- GRU: gi = x_t@Wih^T, gh = weighted@Whh^T; wave w owns {r,z,n} cols 16w..16w+15
    v8f ar = vzero8(), az = vzero8(), ain = vzero8(), ahn = vzero8();
#pragma unroll
    for (int k0 = 0; k0 < DD; k0 += 32) {
      v16bf a = afrag_g32(xb, TD, lm, k0, hi);
      ar  = wmma_bf(a, bfrag(w_wih, DD, 16 * wave, k0, lm, hi), ar);
      az  = wmma_bf(a, bfrag(w_wih, DD, HH + 16 * wave, k0, lm, hi), az);
      ain = wmma_bf(a, bfrag(w_wih, DD, 2 * HH + 16 * wave, k0, lm, hi), ain);
    }
#pragma unroll
    for (int k0 = 0; k0 < HH; k0 += 32) {
      v16bf a = afrag_lds(&wgt_bf[0][0], HH, lm, k0, hi);
      ar  = wmma_bf(a, bfrag(w_whh, HH, 16 * wave, k0, lm, hi), ar);
      az  = wmma_bf(a, bfrag(w_whh, HH, HH + 16 * wave, k0, lm, hi), az);
      ahn = wmma_bf(a, bfrag(w_whh, HH, 2 * HH + 16 * wave, k0, lm, hi), ahn);
    }

    // prefetch next x tile while the gates resolve
    if (t + 1 < TT) __builtin_prefetch(xb + (size_t)lm * TD + DD + hi * 128, 0, 0);

    float nh[8];
#pragma unroll
    for (int v = 0; v < 8; ++v) {
      float rg = sigmoidf_(ar[v] + c_bihr + c_bhhr);
      float zg = sigmoidf_(az[v] + c_bihz + c_bhhz);
      float ng = tanhf(ain[v] + c_bihn + rg * (ahn[v] + c_bhhn));
      float hw = (float)wgt_bf[v + 8 * hi][jc];
      nh[v] = (1.f - zg) * ng + zg * hw;
    }
    __syncthreads();   // everyone done reading ring/wgt/h before overwrite
#pragma unroll
    for (int v = 0; v < 8; ++v) {
      __bf16 hb = (__bf16)nh[v];
      h_bf[v + 8 * hi][jc] = hb;
      ring[wptr][v + 8 * hi][jc] = hb;   // append h_t (used at step t+1)
    }
    __syncthreads();

    // ---- fusion output: out[b,t,:] = concat(h_t, static) @ fus_w^T + fus_b
    v8f fo = vzero8();
#pragma unroll
    for (int k0 = 0; k0 < HH; k0 += 32)
      fo = wmma_bf(afrag_lds(&h_bf[0][0], HH, lm, k0, hi),
                   bfrag(w_fus, HH + SS, 16 * wave, k0, lm, hi), fo);
#pragma unroll
    for (int k0 = HH; k0 < HH + SS; k0 += 32)
      fo = wmma_bf(afrag_g32(statb, SS, lm, k0 - HH, hi),
                   bfrag(w_fus, HH + SS, 16 * wave, k0, lm, hi), fo);
#pragma unroll
    for (int v = 0; v < 8; ++v)
      out[(size_t)(b0 + v + 8 * hi) * TT * HH + (size_t)t * HH + jc] = fo[v] + c_fusb;

    wptr = (wptr + 1 == AAC) ? 0 : wptr + 1;
  }
}

extern "C" void kernel_launch(void* const* d_in, const int* in_sizes, int n_in,
                              void* d_out, int out_size, void* d_ws, size_t ws_size,
                              hipStream_t stream) {
  (void)in_sizes; (void)n_in; (void)out_size; (void)ws_size;
  const float* x     = (const float*)d_in[0];
  // d_in[1] = lengths (unused by the reference forward body)
  const float* stat  = (const float*)d_in[2];
  const float* a1_w1 = (const float*)d_in[3];
  const float* a1_b1 = (const float*)d_in[4];
  const float* a2_w1 = (const float*)d_in[5];
  const float* a2_b1 = (const float*)d_in[6];
  const float* a1_w2 = (const float*)d_in[7];
  const float* a1_b2 = (const float*)d_in[8];
  const float* a2_w2 = (const float*)d_in[9];
  const float* a2_b2 = (const float*)d_in[10];
  const float* wih   = (const float*)d_in[11];
  const float* whh   = (const float*)d_in[12];
  const float* bih   = (const float*)d_in[13];
  const float* bhh   = (const float*)d_in[14];
  const float* initw = (const float*)d_in[15];
  const float* initb = (const float*)d_in[16];
  const float* fusw  = (const float*)d_in[17];
  const float* fusb  = (const float*)d_in[18];

  __bf16* wsb = (__bf16*)d_ws;
  prep_weights<<<(WS_TOTAL + 255) / 256, 256, 0, stream>>>(
      wih, whh, a1_w1, a2_w1, a1_w2, a2_w2, initw, fusw, wsb);
  agent_scan<<<BB / 16, 256, 0, stream>>>(
      x, stat, a1_b1, a2_b1, a1_b2, a2_b2, bih, bhh, initb, fusb,
      wsb, (float*)d_out);
}
